// Quantizer_23407571763538
// MI455X (gfx1250) — matrix operand order repaired and checked by reading, not compiled, not run on hardware
//
#include <hip/hip_runtime.h>
#include <hip/hip_bf16.h>

// ---------------------------------------------------------------------------
// VQ quantizer for MI455X (gfx1250, wave32, WMMA).
// B=16, T=2048, D=256, K=4096.  N = B*T = 32768 tokens.
// Outputs (concatenated in d_out as floats):
//   [0]                  c_loss
//   [1 .. 1+N*D)         quantized (fp32 gather of codebook rows)
//   [1+N*D .. +N)        encoding_index (int32 bit patterns)
// ---------------------------------------------------------------------------

#define VQ_D   256
#define VQ_K   4096
#define VQ_N   (16 * 2048)         // 32768 tokens
#define VQ_TILES (VQ_N / 16)       // 2048 16-token tiles
#define VQ_BLOCKS (VQ_TILES / 8)   // 256 blocks, 8 waves/block

typedef __attribute__((ext_vector_type(16))) __bf16 v16bf;
typedef __attribute__((ext_vector_type(8)))  float  v8f;

// ---------------------------------------------------------------------------
// Kernel 1: codebook fp32 -> bf16, plus halfNorm[k] = 0.5*sum(e_k^2).
// One wave per codebook row; 8 rows per 256-thread block; 512 blocks.
// ---------------------------------------------------------------------------
__global__ __launch_bounds__(256)
void vq_prep_kernel(const float* __restrict__ emb,
                    __bf16* __restrict__ embB,
                    float* __restrict__ halfNorm) {
    const int lane = threadIdx.x & 31;
    const int wave = threadIdx.x >> 5;
    const int row  = blockIdx.x * 8 + wave;           // 0..4095
    const float*  r = emb  + (size_t)row * VQ_D;
    __bf16*       w = embB + (size_t)row * VQ_D;
    float s = 0.0f;
#pragma unroll
    for (int i = 0; i < VQ_D / 32; ++i) {
        float v = r[lane + i * 32];
        s += v * v;
        w[lane + i * 32] = (__bf16)v;
    }
#pragma unroll
    for (int sh = 16; sh >= 1; sh >>= 1) s += __shfl_xor(s, sh);
    if (lane == 0) halfNorm[row] = 0.5f * s;
}

// ---------------------------------------------------------------------------
// Kernel 2: per-wave 16-token tile.  A (16x256 bf16 tokens) lives in VGPRs,
// stream codebook in 16-code chunks with v_wmma_f32_16x16x32_bf16, track
// running argmin of (0.5*||e||^2 - x.e), then gather fp32 rows + loss partial.
// ---------------------------------------------------------------------------
__global__ __launch_bounds__(256)
void vq_argmin_kernel(const float* __restrict__ tok,     // [N, D] fp32
                      const float* __restrict__ emb,     // [K, D] fp32
                      const __bf16* __restrict__ embB,   // [K, D] bf16
                      const float* __restrict__ halfNorm,// [K]
                      float* __restrict__ lossPartial,   // [gridDim.x]
                      float* __restrict__ outQ,          // [N, D] (d_out+1)
                      int*   __restrict__ outIdx) {      // [N]
    const int lane = threadIdx.x & 31;
    const int wave = threadIdx.x >> 5;
    const int col  = lane & 15;                 // C column / A row / B column
    const bool hi  = lane >= 16;
    const int tile = blockIdx.x * 8 + wave;     // 0..2047
    const int tokBase = tile * 16;

    // ---- Load + convert A fragments: token row m=col, all 256 dims ----
    // A layout (matches contiguous B): lo lanes elems 0..15 <- K{0..15};
    // hi lanes elems 0..15 <- K{16..31}.
    const float* trow = tok + (size_t)(tokBase + col) * VQ_D;
    v16bf a[8];
#pragma unroll
    for (int f = 0; f < 8; ++f) {
        const int d0 = f * 32 + (hi ? 16 : 0);
        const int d1 = f * 32 + (hi ? 24 : 8);
        float4 x0 = *(const float4*)(trow + d0);
        float4 x1 = *(const float4*)(trow + d0 + 4);
        float4 x2 = *(const float4*)(trow + d1);
        float4 x3 = *(const float4*)(trow + d1 + 4);
        a[f][0]  = (__bf16)x0.x; a[f][1]  = (__bf16)x0.y;
        a[f][2]  = (__bf16)x0.z; a[f][3]  = (__bf16)x0.w;
        a[f][4]  = (__bf16)x1.x; a[f][5]  = (__bf16)x1.y;
        a[f][6]  = (__bf16)x1.z; a[f][7]  = (__bf16)x1.w;
        a[f][8]  = (__bf16)x2.x; a[f][9]  = (__bf16)x2.y;
        a[f][10] = (__bf16)x2.z; a[f][11] = (__bf16)x2.w;
        a[f][12] = (__bf16)x3.x; a[f][13] = (__bf16)x3.y;
        a[f][14] = (__bf16)x3.z; a[f][15] = (__bf16)x3.w;
    }

    // ---- Running argmin over K, 16 codes per chunk ----
    float minv[8];
    int   mini[8];
#pragma unroll
    for (int j = 0; j < 8; ++j) { minv[j] = 3.402823466e38f; mini[j] = 0; }

#pragma unroll 1
    for (int k0 = 0; k0 < VQ_K; k0 += 16) {
        // B 32x16 bf16 layout: lane = column n (code k0+col), 16 contiguous
        // K values, hi half offset +16 -> one contiguous 32B load per lane.
        const __bf16* bbase =
            embB + (size_t)(k0 + col) * VQ_D + (hi ? 16 : 0);
        v8f c = {};
#pragma unroll
        for (int f = 0; f < 8; ++f) {
            v16bf b = *(const v16bf*)(bbase + f * 32);
            c = __builtin_amdgcn_wmma_f32_16x16x32_bf16(
                    false, a[f], false, b, (short)0, c, false, false);
        }
        const float h = halfNorm[k0 + col];      // 0.5*||e||^2, same argmin
#pragma unroll
        for (int j = 0; j < 8; ++j) {
            float v = h - c[j];                  // ~ 0.5*d2 - 0.5*||x||^2
            if (v < minv[j]) { minv[j] = v; mini[j] = k0 + col; }
        }
    }

    // ---- Reduce argmin across the 16 lanes sharing each C row ----
#pragma unroll
    for (int j = 0; j < 8; ++j) {
        float mv = minv[j]; int mi = mini[j];
#pragma unroll
        for (int s = 8; s >= 1; s >>= 1) {
            float ov = __shfl_xor(mv, s);
            int   oi = __shfl_xor(mi, s);
            if (ov < mv || (ov == mv && oi < mi)) { mv = ov; mi = oi; }
        }
        minv[j] = mv; mini[j] = mi;
    }

    // Broadcast the 16 winning indices to the whole wave.
    int idxs[16];
#pragma unroll
    for (int j = 0; j < 8; ++j) {
        idxs[j]     = __shfl(mini[j], 0);    // rows 0..7 live in lo half
        idxs[j + 8] = __shfl(mini[j], 16);   // rows 8..15 in hi half
    }

    // ---- Gather fp32 code rows, write quantized, accumulate loss ----
    float lsum = 0.0f;
#pragma unroll
    for (int m = 0; m < 16; ++m) {
        const int e = idxs[m];
        const float* erow = emb + (size_t)e * VQ_D;
        const float* xrow = tok + (size_t)(tokBase + m) * VQ_D;
        float* qrow = outQ + (size_t)(tokBase + m) * VQ_D;
#pragma unroll
        for (int c4 = 0; c4 < 2; ++c4) {
            const int d = lane * 4 + c4 * 128;
            float4 ev = *(const float4*)(erow + d);
            float4 xv = *(const float4*)(xrow + d);
            // outQ base is d_out+1 -> 4-mod-16 aligned: scalar stores.
            qrow[d + 0] = ev.x; qrow[d + 1] = ev.y;
            qrow[d + 2] = ev.z; qrow[d + 3] = ev.w;
            float dx = ev.x - xv.x, dy = ev.y - xv.y;
            float dz = ev.z - xv.z, dw = ev.w - xv.w;
            lsum += dx * dx + dy * dy + dz * dz + dw * dw;
        }
        if (lane == 0) outIdx[tokBase + m] = e;
    }

    // Deterministic block partial for the loss (no float atomics).
#pragma unroll
    for (int s = 16; s >= 1; s >>= 1) lsum += __shfl_xor(lsum, s);
    __shared__ float red[8];
    if (lane == 0) red[wave] = lsum;
    __syncthreads();
    if (threadIdx.x == 0) {
        float t = 0.0f;
#pragma unroll
        for (int w = 0; w < 8; ++w) t += red[w];
        lossPartial[blockIdx.x] = t;
    }
}

// ---------------------------------------------------------------------------
// Kernel 3: reduce 256 block partials -> c_loss.
// ---------------------------------------------------------------------------
__global__ __launch_bounds__(256)
void vq_loss_kernel(const float* __restrict__ partial,
                    float* __restrict__ outLoss) {
    __shared__ float sm[256];
    sm[threadIdx.x] = partial[threadIdx.x];
    __syncthreads();
    for (int s = 128; s > 0; s >>= 1) {
        if (threadIdx.x < s) sm[threadIdx.x] += sm[threadIdx.x + s];
        __syncthreads();
    }
    if (threadIdx.x == 0)
        outLoss[0] = 0.5f * sm[0] / (float)((size_t)VQ_N * VQ_D);
}

// ---------------------------------------------------------------------------
extern "C" void kernel_launch(void* const* d_in, const int* in_sizes, int n_in,
                              void* d_out, int out_size, void* d_ws, size_t ws_size,
                              hipStream_t stream) {
    const float* tok = (const float*)d_in[0];  // [B,T,D] fp32
    const float* emb = (const float*)d_in[1];  // [K,D]   fp32

    float* out     = (float*)d_out;
    float* outLoss = out;
    float* outQ    = out + 1;
    int*   outIdx  = (int*)(out + 1 + (size_t)VQ_N * VQ_D);

    // Workspace layout: bf16 codebook (2MB) | halfNorm (16KB) | partials (1KB)
    __bf16* embB     = (__bf16*)d_ws;
    float*  halfNorm = (float*)((char*)d_ws + (size_t)VQ_K * VQ_D * sizeof(__bf16));
    float*  partial  = halfNorm + VQ_K;

    vq_prep_kernel<<<VQ_K / 8, 256, 0, stream>>>(emb, embB, halfNorm);
    vq_argmin_kernel<<<VQ_BLOCKS, 256, 0, stream>>>(
        tok, emb, embB, halfNorm, partial, outQ, outIdx);
    vq_loss_kernel<<<1, 256, 0, stream>>>(partial, outLoss);
}
